// DeformableConv2D_49435073577767
// MI455X (gfx1250) — matrix operand order repaired and checked by reading, not compiled, not run on hardware
//
#include <hip/hip_runtime.h>

typedef __attribute__((ext_vector_type(16))) __bf16 v16bf;
typedef __attribute__((ext_vector_type(8)))  __bf16 v8bf;
typedef __attribute__((ext_vector_type(4)))  __bf16 v4bf;
typedef __attribute__((ext_vector_type(8)))  float  v8f;
typedef __attribute__((ext_vector_type(4)))  int    v4i;

#define BN 8
#define HH 128
#define WW 128
#define CC 64
#define FF 128
#define KTAPS 9
#define CS 576                  // KTAPS*CC, sampled channels
#define NPIX (BN*HH*WW)         // 131072
#define MTILE 32                // pixels per block
#define ASTRIDE 600             // LDS row stride in halves (bank-conflict free)

#define WT_ELEMS (KTAPS*FF*CS)  // 663552

// ws layout (bytes):
//   [0)        offsets     f32  NPIX*18
//   [OFF_WHI)  Wt_hi       bf16 WT_ELEMS   (layout [t][n][kc])
//   [OFF_WLO)  Wt_lo       bf16 WT_ELEMS
//   [OFF_SHI)  sampled_hi  bf16 NPIX*CS
//   [OFF_SLO)  sampled_lo  bf16 NPIX*CS
#define OFF_WHI ((size_t)NPIX * 18 * 4)
#define OFF_WLO (OFF_WHI + (size_t)WT_ELEMS * 2)
#define OFF_SHI (OFF_WLO + (size_t)WT_ELEMS * 2)
#define OFF_SLO (OFF_SHI + (size_t)NPIX * CS * 2)

#define HAS_ASYNC_LDS __has_builtin(__builtin_amdgcn_global_load_async_to_lds_b128)

__device__ __forceinline__ void split_bf16(float x, __bf16& hi, __bf16& lo) {
    hi = (__bf16)x;
    lo = (__bf16)(x - (float)hi);
}

// ---------------------------------------------------------------------------
// Repack W (HWIO [t][kc][n]) -> [t][n][kc], split into bf16 hi/lo pair.
__global__ void transpose_split_w(const float* __restrict__ W,
                                  __bf16* __restrict__ Whi,
                                  __bf16* __restrict__ Wlo) {
    int idx = blockIdx.x * 256 + threadIdx.x;
    if (idx >= WT_ELEMS) return;
    int t  = idx / (CS * FF);
    int r  = idx - t * (CS * FF);
    int kc = r / FF;
    int n  = r - kc * FF;
    float v = W[idx];
    __bf16 hi, lo;
    split_bf16(v, hi, lo);
    size_t o = (size_t)(t * FF + n) * CS + kc;
    Whi[o] = hi;
    Wlo[o] = lo;
}

// ---------------------------------------------------------------------------
// Offset conv: 3x3 x 64 -> 18 per pixel (1.5% of FLOPs, plain VALU).
__global__ void offset_conv(const float* __restrict__ x,
                            const float* __restrict__ Woff,
                            const float* __restrict__ boff,
                            float* __restrict__ offs) {
    int p = blockIdx.x * 256 + threadIdx.x;
    if (p >= NPIX) return;
    int b = p >> 14;
    int h = (p >> 7) & 127;
    int w = p & 127;

    float acc[18];
#pragma unroll
    for (int j = 0; j < 18; ++j) acc[j] = boff[j];

    for (int ky = 0; ky < 3; ++ky) {
        int y = h + ky - 1;
        if ((unsigned)y >= HH) continue;
        for (int kx = 0; kx < 3; ++kx) {
            int xx = w + kx - 1;
            if ((unsigned)xx >= WW) continue;
            const float* xrow = x + ((size_t)((b * HH + y) * WW + xx)) * CC;
            const float* wrow = Woff + (size_t)((ky * 3 + kx) * CC) * 18;
            for (int c = 0; c < CC; ++c) {
                float xv = xrow[c];
                const float* wc = wrow + c * 18;
#pragma unroll
                for (int j = 0; j < 18; ++j) acc[j] = fmaf(xv, wc[j], acc[j]);
            }
        }
    }
    float* o = offs + (size_t)p * 18;
#pragma unroll
    for (int j = 0; j < 18; ++j) o[j] = acc[j];
}

// ---------------------------------------------------------------------------
// Bilinear sampling: one thread per (pixel, tap); 4-corner float4 gathers.
// Emits the sampled map pre-split into bf16 hi/lo (consumed by WMMA GEMM).
__global__ void bilinear_sample(const float* __restrict__ x,
                                const float* __restrict__ offs,
                                __bf16* __restrict__ Shi,
                                __bf16* __restrict__ Slo) {
    int pt = blockIdx.x * 256 + threadIdx.x;
    if (pt >= NPIX * KTAPS) return;
    int p   = pt / KTAPS;
    int tap = pt - p * KTAPS;
    int b = p >> 14;
    int h = (p >> 7) & 127;
    int w = p & 127;

    float offx = offs[(size_t)p * 18 + tap * 2 + 0];
    float offy = offs[(size_t)p * 18 + tap * 2 + 1];
    float lx = (float)w + (float)(tap % 3 - 1) + offx;
    float ly = (float)h + (float)(tap / 3 - 1) + offy;
    lx = fminf(fmaxf(lx, 0.f), (float)(WW - 1));
    ly = fminf(fmaxf(ly, 0.f), (float)(HH - 1));

    float x0 = floorf(lx);
    float y0 = floorf(ly);
    float x1 = fminf(x0 + 1.f, (float)(WW - 1));
    float y1 = fminf(y0 + 1.f, (float)(HH - 1));

    float wa = (x1 - lx) * (y1 - ly);
    float wb = (x1 - lx) * (ly - y0);
    float wc = (lx - x0) * (y1 - ly);
    float wd = (lx - x0) * (ly - y0);

    int x0i = (int)x0, x1i = (int)x1, y0i = (int)y0, y1i = (int)y1;
    const float* Ia = x + ((size_t)((b * HH + y0i) * WW + x0i)) * CC;
    const float* Ib = x + ((size_t)((b * HH + y1i) * WW + x0i)) * CC;
    const float* Ic = x + ((size_t)((b * HH + y0i) * WW + x1i)) * CC;
    const float* Id = x + ((size_t)((b * HH + y1i) * WW + x1i)) * CC;
    __bf16* outH = Shi + (size_t)p * CS + tap * CC;
    __bf16* outL = Slo + (size_t)p * CS + tap * CC;

#pragma unroll
    for (int c = 0; c < CC; c += 4) {
        float4 a4 = *(const float4*)(Ia + c);
        float4 b4 = *(const float4*)(Ib + c);
        float4 c4 = *(const float4*)(Ic + c);
        float4 d4 = *(const float4*)(Id + c);
        float4 r;
        r.x = wa * a4.x + wb * b4.x + wc * c4.x + wd * d4.x;
        r.y = wa * a4.y + wb * b4.y + wc * c4.y + wd * d4.y;
        r.z = wa * a4.z + wb * b4.z + wc * c4.z + wd * d4.z;
        r.w = wa * a4.w + wb * b4.w + wc * c4.w + wd * d4.w;
        v4bf rh, rl;
        __bf16 hi, lo;
        split_bf16(r.x, hi, lo); rh[0] = hi; rl[0] = lo;
        split_bf16(r.y, hi, lo); rh[1] = hi; rl[1] = lo;
        split_bf16(r.z, hi, lo); rh[2] = hi; rl[2] = lo;
        split_bf16(r.w, hi, lo); rh[3] = hi; rl[3] = lo;
        *(v4bf*)(outH + c) = rh;
        *(v4bf*)(outL + c) = rl;
    }
}

// ---------------------------------------------------------------------------
// Main conv as implicit GEMM, fp32 emulated via 3x BF16 WMMA (split hi/lo).
// Block: 256 threads = 8 wave32. M-tile = 32 pixels; each wave owns a 16-wide
// N-tile and two 16x16 accumulators (B fragments reused across both).
// A tiles staged global->LDS via GLOBAL_LOAD_ASYNC_TO_LDS_B128 when available.
__global__ __launch_bounds__(256)
void deform_gemm(const __bf16* __restrict__ Shi,
                 const __bf16* __restrict__ Slo,
                 const __bf16* __restrict__ Whi,
                 const __bf16* __restrict__ Wlo,
                 const float* __restrict__ bias,
                 float* __restrict__ out) {
    __shared__ __bf16 Ahi[MTILE * ASTRIDE];
    __shared__ __bf16 Alo[MTILE * ASTRIDE];

    const int tid  = threadIdx.x;
    const int lane = tid & 31;
    const int wave = tid >> 5;
    const int m0 = blockIdx.x * MTILE;   // tile stays within one image row
    const int n0 = wave * 16;
    const int h  = (m0 >> 7) & 127;
    const int w0 = m0 & 127;

    const int row  = lane & 15;
    const int selA = (lane < 16) ? 0 : 8;    // 16-bit A frag: K = k+sel..+7, k+16+sel..+23
    const int selB = (lane < 16) ? 0 : 8;    // 16-bit B frag: K = k+sel..+7, k+16+sel..+23

    v8f acc0 = {0.f, 0.f, 0.f, 0.f, 0.f, 0.f, 0.f, 0.f};
    v8f acc1 = {0.f, 0.f, 0.f, 0.f, 0.f, 0.f, 0.f, 0.f};

    union V16 { v16bf v; v8bf h[2]; };

    for (int t = 0; t < KTAPS; ++t) {
        const int dy = t / 3 - 1;
        const int dx = t % 3 - 1;
        __syncthreads();
        const int yq = h + dy;
        const bool yok = (unsigned)yq < HH;
        // stage 32 x 576 bf16 hi/lo slices as 16B chunks: 32 rows x 72 chunks
        for (int idx = tid; idx < MTILE * 72; idx += 256) {
            const int i = idx / 72;
            const int c = (idx - i * 72) * 8;          // half-element offset
            const int xq = w0 + i + dx;
            const int ldsOff = i * ASTRIDE + c;
            if (yok && (unsigned)xq < WW) {
                const int q = m0 + i + dy * WW + dx;
                const size_t g = (size_t)q * CS + c;
#if HAS_ASYNC_LDS
                __builtin_amdgcn_global_load_async_to_lds_b128(
                    (__attribute__((address_space(1))) v4i*)(v4i*)(Shi + g),
                    (__attribute__((address_space(3))) v4i*)(v4i*)(&Ahi[ldsOff]), 0, 0);
                __builtin_amdgcn_global_load_async_to_lds_b128(
                    (__attribute__((address_space(1))) v4i*)(v4i*)(Slo + g),
                    (__attribute__((address_space(3))) v4i*)(v4i*)(&Alo[ldsOff]), 0, 0);
#else
                *(v8bf*)(&Ahi[ldsOff]) = *(const v8bf*)(Shi + g);
                *(v8bf*)(&Alo[ldsOff]) = *(const v8bf*)(Slo + g);
#endif
            } else {
                v8bf z = {};
                *(v8bf*)(&Ahi[ldsOff]) = z;
                *(v8bf*)(&Alo[ldsOff]) = z;
            }
        }
#if HAS_ASYNC_LDS
#if __has_builtin(__builtin_amdgcn_s_wait_asynccnt)
        __builtin_amdgcn_s_wait_asynccnt(0);
#else
        asm volatile("s_wait_asynccnt 0x0" ::: "memory");
#endif
#endif
        __syncthreads();

        const __bf16* bhrow = Whi + (size_t)(t * FF + n0 + row) * CS;
        const __bf16* blrow = Wlo + (size_t)(t * FF + n0 + row) * CS;
#pragma unroll 2
        for (int k = 0; k < CS; k += 32) {
            V16 bhu, blu;
            bhu.h[0] = *(const v8bf*)(bhrow + k + selB);   bhu.h[1] = *(const v8bf*)(bhrow + k + selB + 16);
            blu.h[0] = *(const v8bf*)(blrow + k + selB);   blu.h[1] = *(const v8bf*)(blrow + k + selB + 16);
            v16bf bh = bhu.v;
            v16bf bl = blu.v;

            V16 ah0, al0, ah1, al1;
            const __bf16* a0 = &Ahi[row * ASTRIDE + k + selA];
            const __bf16* l0 = &Alo[row * ASTRIDE + k + selA];
            const __bf16* a1 = &Ahi[(16 + row) * ASTRIDE + k + selA];
            const __bf16* l1 = &Alo[(16 + row) * ASTRIDE + k + selA];
            ah0.h[0] = *(const v8bf*)(a0);      ah0.h[1] = *(const v8bf*)(a0 + 16);
            al0.h[0] = *(const v8bf*)(l0);      al0.h[1] = *(const v8bf*)(l0 + 16);
            ah1.h[0] = *(const v8bf*)(a1);      ah1.h[1] = *(const v8bf*)(a1 + 16);
            al1.h[0] = *(const v8bf*)(l1);      al1.h[1] = *(const v8bf*)(l1 + 16);

            // fp32 emulation: hi*hi + hi*lo + lo*hi, f32 accumulate
            acc0 = __builtin_amdgcn_wmma_f32_16x16x32_bf16(false, ah0.v, false, bh, (short)0, acc0, false, false);
            acc0 = __builtin_amdgcn_wmma_f32_16x16x32_bf16(false, ah0.v, false, bl, (short)0, acc0, false, false);
            acc0 = __builtin_amdgcn_wmma_f32_16x16x32_bf16(false, al0.v, false, bh, (short)0, acc0, false, false);

            acc1 = __builtin_amdgcn_wmma_f32_16x16x32_bf16(false, ah1.v, false, bh, (short)0, acc1, false, false);
            acc1 = __builtin_amdgcn_wmma_f32_16x16x32_bf16(false, ah1.v, false, bl, (short)0, acc1, false, false);
            acc1 = __builtin_amdgcn_wmma_f32_16x16x32_bf16(false, al1.v, false, bh, (short)0, acc1, false, false);
        }
    }

    // C/D layout: VGPR r -> (M = r + (lane<16?0:8), N = lane&15)
    const int colN  = n0 + row;
    const int rbase = (lane < 16) ? 0 : 8;
    const float bv  = bias[colN];
#pragma unroll
    for (int r = 0; r < 8; ++r) {
        out[(size_t)(m0 + rbase + r) * FF + colN] = acc0[r] + bv;
        out[(size_t)(m0 + 16 + rbase + r) * FF + colN] = acc1[r] + bv;
    }
}

// ---------------------------------------------------------------------------
extern "C" void kernel_launch(void* const* d_in, const int* in_sizes, int n_in,
                              void* d_out, int out_size, void* d_ws, size_t ws_size,
                              hipStream_t stream) {
    (void)in_sizes; (void)n_in; (void)out_size; (void)ws_size;
    const float* x     = (const float*)d_in[0];
    const float* Woff  = (const float*)d_in[1];
    const float* boff  = (const float*)d_in[2];
    const float* Wmain = (const float*)d_in[3];
    const float* bias  = (const float*)d_in[4];
    float* out = (float*)d_out;

    char* ws = (char*)d_ws;
    float*  offs = (float*)ws;
    __bf16* Whi  = (__bf16*)(ws + OFF_WHI);
    __bf16* Wlo  = (__bf16*)(ws + OFF_WLO);
    __bf16* Shi  = (__bf16*)(ws + OFF_SHI);
    __bf16* Slo  = (__bf16*)(ws + OFF_SLO);

    transpose_split_w<<<(WT_ELEMS + 255) / 256, 256, 0, stream>>>(Wmain, Whi, Wlo);
    offset_conv<<<(NPIX + 255) / 256, 256, 0, stream>>>(x, Woff, boff, offs);
    bilinear_sample<<<(NPIX * KTAPS + 255) / 256, 256, 0, stream>>>(x, offs, Shi, Slo);
    deform_gemm<<<NPIX / MTILE, 256, 0, stream>>>(Shi, Slo, Whi, Wlo, bias, out);
}
